// CrossAttention_24481313587375
// MI455X (gfx1250) — matrix operand order repaired and checked by reading, not compiled, not run on hardware
//
#include <hip/hip_runtime.h>
#include <hip/hip_bf16.h>

// CDNA5 / gfx1250 shifted-window cross-attention, WMMA bf16 path.
//   b_d[i,j] = (q_i+d).(pe_i - pe_j) = G[i,i] - G[i,j],  G = (q+d) @ pe_h^T
// eliminates the 537MB diff tensor. All operands stored so every WMMA
// fragment is 1-2 contiguous 16/32-byte loads (no scalar gathers):
//   A fragments: two 8x bf16 runs (ISA 7.12.2 K-half split)
//   B fragments: two 8x bf16 runs, same K split (column-major / d-major storage)

typedef __bf16 bf16;
typedef __attribute__((ext_vector_type(16))) __bf16 v16bf;
typedef __attribute__((ext_vector_type(8)))  __bf16 v8bf;
typedef __attribute__((ext_vector_type(8)))  float  v8f;

#define HEADS 8
#define HDIM  256
#define DPH   32
#define WINSZ 64
#define NW    32
#define BATCH 4
#define LSEQ  2048
#define NTOK  (BATCH*LSEQ)   // 8192

__device__ __forceinline__ bf16  f2bf(float x){ return (bf16)x; }
__device__ __forceinline__ float bf2f(bf16 x){ return (float)x; }

// shifted-window permutation (X=2, WIN=64): fused into address math
__device__ __forceinline__ int s2o(int s){ return ((s>>4)&63)*32 + (s>>10)*16 + (s&15); }
__device__ __forceinline__ int o2s(int o){ return ((o>>4)&1)*1024 + (o>>5)*16 + (o&15); }

// element e (0..15) of an A fragment -> K offset (two 8-wide runs)
__device__ __forceinline__ int a_k(int e,int lane){ return ((e>>3)<<4) + ((lane>>4)<<3) + (e&7); }

__device__ __forceinline__ v8f wmma_bf16(v16bf a, v16bf b, v8f c){
    return __builtin_amdgcn_wmma_f32_16x16x32_bf16(false, a, false, b, (short)0, c, false, false);
}

// contiguous 32B load (unused): caller guarantees 32B alignment
__device__ __forceinline__ v16bf ld16(const bf16* p){ return *(const v16bf*)p; }
// two contiguous 16B chunks (A fragments)
__device__ __forceinline__ v16bf ld2x8(const bf16* p0, const bf16* p1){
    v8bf a = *(const v8bf*)p0, b = *(const v8bf*)p1;
    return __builtin_shufflevector(a, b, 0,1,2,3,4,5,6,7,8,9,10,11,12,13,14,15);
}

__device__ __forceinline__ float gelu_exact(float y){
    return 0.5f*y*(1.f + erff(y*0.70710678118654752f));
}

// ---------------------------------------------------------------------------
// Prep 1: transpose+convert the five GEMM weights to bf16 column-major
// Wt[w][n*256+k].  grid=(256 k-rows, 5), block=256 (thread = n, coalesced read)
// ---------------------------------------------------------------------------
__global__ __launch_bounds__(256)
void prep_w_kernel(const float* __restrict__ wq, const float* __restrict__ wk,
                   const float* __restrict__ wv, const float* __restrict__ p2w,
                   const float* __restrict__ wo, bf16* __restrict__ Wt)
{
    const int w = blockIdx.y;
    const float* src = w==0 ? wq : w==1 ? wk : w==2 ? wv : w==3 ? p2w : wo;
    const int k = blockIdx.x, n = threadIdx.x;
    Wt[(size_t)w*HDIM*HDIM + (size_t)n*HDIM + k] = f2bf(src[(size_t)k*HDIM + n]);
}

// ---------------------------------------------------------------------------
// Prep 2: inputs -> bf16 in SHIFTED row order (one conversion for all 12
// QKV column-tiles).  grid=NTOK, block=256.
// ---------------------------------------------------------------------------
__global__ __launch_bounds__(256)
void xshift_kernel(const float* __restrict__ x, bf16* __restrict__ Xs)
{
    const int s = blockIdx.x, n = threadIdx.x;
    const int b = s >> 11;
    const int o = (b<<11) + s2o(s & 2047);
    Xs[(size_t)s*HDIM + n] = f2bf(x[(size_t)o*HDIM + n]);
}

// ---------------------------------------------------------------------------
// Kernel 1: fused QKV projection GEMM from shifted bf16 activations.
// Q/K tiles stored (i,d); V tile stored TRANSPOSED (d,i) so the attn@v
// B-operand is two contiguous 16B loads.
// grid = (NTOK/64, 12): y = proj*4 + 64-col tile.  4 waves, 16-row strip.
// ---------------------------------------------------------------------------
__global__ __launch_bounds__(128)
void qkv_kernel(const bf16* __restrict__ Xs, const bf16* __restrict__ Wt,
                const float* __restrict__ bq, const float* __restrict__ bk,
                const float* __restrict__ bv,
                bf16* __restrict__ Q, bf16* __restrict__ Kd, bf16* __restrict__ V)
{
    const int lane = threadIdx.x & 31, wid = threadIdx.x >> 5;
    const int proj = blockIdx.y >> 2, n64 = (blockIdx.y & 3) * 64;
    const bf16*  W  = Wt + (size_t)proj*HDIM*HDIM;
    const float* Bb = proj==0 ? bq : (proj==1 ? bk : bv);
    bf16* Out = proj==0 ? Q : (proj==1 ? Kd : V);

    const int khalf = (lane>>4)<<3;          // A-fragment K-half offset
    const int kb16  = (lane>>4)<<4;          // unused; B fragments use khalf
    const int srow  = blockIdx.x*64 + wid*16 + (lane & 15);
    const bf16* arow = Xs + (size_t)srow * HDIM;

    v8f acc[4];
    for (int nt=0; nt<4; ++nt) for (int e=0; e<8; ++e) acc[nt][e] = 0.f;

    for (int kk = 0; kk < HDIM; kk += 32) {
        v16bf a = ld2x8(arow + kk + khalf, arow + kk + 16 + khalf);
        #pragma unroll
        for (int nt=0; nt<4; ++nt) {
            int nc = n64 + nt*16 + (lane & 15);
            v16bf bm = ld2x8(W + (size_t)nc*HDIM + kk + khalf, W + (size_t)nc*HDIM + kk + 16 + khalf);
            acc[nt] = wmma_bf16(a, bm, acc[nt]);
        }
    }

    const float qscale = 0.17677669529663687f;   // 1/sqrt(32)
    #pragma unroll
    for (int nt=0; nt<4; ++nt) {
        int nc = n64 + nt*16 + (lane & 15);
        #pragma unroll
        for (int e=0; e<8; ++e) {
            int sr = blockIdx.x*64 + wid*16 + e + ((lane>>4)<<3);
            float v = acc[nt][e] + Bb[nc];
            if (proj == 0) v *= qscale;
            int b2 = sr >> 11, sl = sr & 2047, win = sl >> 6, i = sl & 63;
            size_t tile = (((size_t)(b2*NW + win))*HEADS + (nc>>5)) * (WINSZ*DPH);
            size_t dst = (proj == 2) ? tile + (size_t)(nc&31)*WINSZ + i    // V: d-major
                                     : tile + (size_t)i*DPH + (nc&31);     // Q/K: i-major
            Out[dst] = f2bf(v);
        }
    }
}

// ---------------------------------------------------------------------------
// Kernel 2a: position MLP stage 1: Linear(2->256) + LayerNorm + GELU.
// ---------------------------------------------------------------------------
__global__ __launch_bounds__(256)
void posmlp1_kernel(const float* __restrict__ p,
                    const float* __restrict__ p1w, const float* __restrict__ p1b,
                    const float* __restrict__ lng, const float* __restrict__ lnb,
                    bf16* __restrict__ Hid)
{
    __shared__ float red[256];
    __shared__ float stat[2];
    const int s = blockIdx.x, n = threadIdx.x;
    const int b = s >> 11;
    const int o = (b<<11) + s2o(s & 2047);
    const float p0 = p[(size_t)o*2+0], p1v = p[(size_t)o*2+1];
    float h = p0*p1w[n] + p1v*p1w[HDIM+n] + p1b[n];
    red[n] = h; __syncthreads();
    for (int off=128; off>0; off>>=1){ if(n<off) red[n]+=red[n+off]; __syncthreads(); }
    if (n==0) stat[0] = red[0]*(1.f/HDIM);
    __syncthreads();
    const float mu = stat[0], dvv = h - mu;
    red[n] = dvv*dvv; __syncthreads();
    for (int off=128; off>0; off>>=1){ if(n<off) red[n]+=red[n+off]; __syncthreads(); }
    if (n==0) stat[1] = red[0]*(1.f/HDIM);
    __syncthreads();
    float y = dvv * rsqrtf(stat[1] + 1e-6f) * lng[n] + lnb[n];
    Hid[(size_t)s*HDIM + n] = f2bf(gelu_exact(y));
}

// ---------------------------------------------------------------------------
// Kernel 2b: pe = GELU(hidden @ p2w + p2b), head-major window-tile store.
// ---------------------------------------------------------------------------
__global__ __launch_bounds__(128)
void pe_kernel(const bf16* __restrict__ Hid, const bf16* __restrict__ Wt,
               const float* __restrict__ p2b, bf16* __restrict__ PE)
{
    const int lane = threadIdx.x & 31, wid = threadIdx.x >> 5;
    const int n64 = blockIdx.y * 64;
    const bf16* W = Wt + (size_t)3*HDIM*HDIM;        // p2w slot
    const int khalf = (lane>>4)<<3, kb16 = (lane>>4)<<4;
    const int srow = blockIdx.x*64 + wid*16 + (lane & 15);
    const bf16* arow = Hid + (size_t)srow * HDIM;

    v8f acc[4];
    for (int nt=0;nt<4;++nt) for(int e=0;e<8;++e) acc[nt][e]=0.f;

    for (int kk=0; kk<HDIM; kk+=32) {
        v16bf a = ld2x8(arow + kk + khalf, arow + kk + 16 + khalf);
        #pragma unroll
        for (int nt=0;nt<4;++nt) {
            int nc = n64 + nt*16 + (lane&15);
            v16bf bm = ld2x8(W + (size_t)nc*HDIM + kk + khalf, W + (size_t)nc*HDIM + kk + 16 + khalf);
            acc[nt] = wmma_bf16(a, bm, acc[nt]);
        }
    }
    #pragma unroll
    for (int nt=0;nt<4;++nt) {
        int nc = n64 + nt*16 + (lane&15);
        #pragma unroll
        for (int e=0;e<8;++e) {
            int sr = blockIdx.x*64 + wid*16 + e + ((lane>>4)<<3);
            float g = gelu_exact(acc[nt][e] + p2b[nc]);
            int b2 = sr >> 11, sl = sr & 2047, win = sl >> 6, i = sl & 63;
            size_t dst = ((((size_t)(b2*NW + win))*HEADS + (nc>>5))*WINSZ + i)*DPH + (nc&31);
            PE[dst] = f2bf(g);
        }
    }
}

// ---------------------------------------------------------------------------
// Kernel 3: per-(b,window,head) attention tile.  1024 blocks, 4 waves.
// A = (q+c) k^T and G = (q+d) pe^T via WMMA; scores = A + diag(G) - G;
// softmax in LDS; ctx = attn @ v via WMMA (V tile is d-major).
// ---------------------------------------------------------------------------
__global__ __launch_bounds__(128)
void attn_kernel(const bf16* __restrict__ Q, const bf16* __restrict__ Kd,
                 const bf16* __restrict__ V, const bf16* __restrict__ PE,
                 const float* __restrict__ cvec, const float* __restrict__ dvec,
                 bf16* __restrict__ CTX)
{
    __shared__ float sA[WINSZ*WINSZ];
    __shared__ float sG[WINSZ*WINSZ];
    __shared__ bf16  sP[WINSZ*WINSZ];
    const int blk = blockIdx.x;                 // b*NW*HEADS + w*HEADS + h
    const int head = blk & (HEADS-1);
    const size_t base = (size_t)blk * WINSZ * DPH;
    const bf16* qT  = Q  + base;
    const bf16* kT  = Kd + base;
    const bf16* vT  = V  + base;                // (d, j) transposed tile
    const bf16* peT = PE + base;
    const int lane = threadIdx.x & 31, wid = threadIdx.x >> 5;
    const int khalf = (lane>>4)<<3, kb16 = (lane>>4)<<4;
    const int mrow = wid*16 + (lane & 15);

    // A operands: q rows + c / + d  (wide load, then elementwise bias)
    v16bf qf = ld2x8(qT + mrow*DPH + khalf, qT + mrow*DPH + 16 + khalf);
    v16bf aq_c, aq_d;
    #pragma unroll
    for (int e=0;e<16;++e) {
        int kidx = a_k(e,lane);
        float qv = bf2f(qf[e]);
        aq_c[e] = f2bf(qv + cvec[head*DPH + kidx]);
        aq_d[e] = f2bf(qv + dvec[head*DPH + kidx]);
    }
    v8f accA[4], accG[4];
    for (int nt=0;nt<4;++nt) for(int e=0;e<8;++e){ accA[nt][e]=0.f; accG[nt][e]=0.f; }
    #pragma unroll
    for (int nt=0;nt<4;++nt) {
        int j = nt*16 + (lane&15);
        v16bf bkf = ld2x8(kT  + j*DPH + khalf, kT  + j*DPH + 16 + khalf);   // B[k][j]=k[j][k]: two 8-wide runs
        v16bf bpf = ld2x8(peT + j*DPH + khalf, peT + j*DPH + 16 + khalf);
        accA[nt] = wmma_bf16(aq_c, bkf, accA[nt]);
        accG[nt] = wmma_bf16(aq_d, bpf, accG[nt]);
    }
    #pragma unroll
    for (int nt=0;nt<4;++nt) {
        #pragma unroll
        for (int e=0;e<8;++e) {
            int i = wid*16 + e + ((lane>>4)<<3);
            int j = nt*16 + (lane&15);
            sA[i*WINSZ+j] = accA[nt][e];
            sG[i*WINSZ+j] = accG[nt][e];
        }
    }
    __syncthreads();
    if (threadIdx.x < WINSZ) {                 // one softmax row per thread
        const int i = threadIdx.x;
        const float gd = sG[i*WINSZ+i];        // diag(G): the pe_i term
        float mx = -1e30f;
        for (int j=0;j<WINSZ;++j) {
            float sc = sA[i*WINSZ+j] + gd - sG[i*WINSZ+j];
            sA[i*WINSZ+j] = sc;
            mx = fmaxf(mx, sc);
        }
        float sum = 0.f;
        for (int j=0;j<WINSZ;++j) {
            float ex = __expf(sA[i*WINSZ+j] - mx);
            sum += ex;
            sA[i*WINSZ+j] = ex;
        }
        float inv = 1.f / sum;
        for (int j=0;j<WINSZ;++j) sP[i*WINSZ+j] = f2bf(sA[i*WINSZ+j]*inv);
    }
    __syncthreads();

    v8f accC[2];
    for (int nt=0;nt<2;++nt) for(int e=0;e<8;++e) accC[nt][e]=0.f;
    #pragma unroll
    for (int ks=0; ks<WINSZ; ks+=32) {
        v16bf af = ld2x8(&sP[mrow*WINSZ + ks + khalf],
                         &sP[mrow*WINSZ + ks + 16 + khalf]);
        #pragma unroll
        for (int nt=0;nt<2;++nt) {
            int dc = nt*16 + (lane&15);
            v16bf bfv = ld2x8(vT + dc*WINSZ + ks + khalf, vT + dc*WINSZ + ks + 16 + khalf);   // d-major: two 8-wide runs
            accC[nt] = wmma_bf16(af, bfv, accC[nt]);
        }
    }
    #pragma unroll
    for (int nt=0;nt<2;++nt) {
        #pragma unroll
        for (int e=0;e<8;++e) {
            int i = wid*16 + e + ((lane>>4)<<3);
            int dc = nt*16 + (lane&15);
            CTX[base + (size_t)i*DPH + dc] = f2bf(accC[nt][e]);
        }
    }
}

// ---------------------------------------------------------------------------
// Kernel 4: output projection GEMM; inverse shift permutation fused into the
// A-operand gather from the window-tile ctx layout.  f32 output + bias.
// ---------------------------------------------------------------------------
__global__ __launch_bounds__(128)
void outproj_kernel(const bf16* __restrict__ CTX, const bf16* __restrict__ Wt,
                    const float* __restrict__ bo, float* __restrict__ out)
{
    const int lane = threadIdx.x & 31, wid = threadIdx.x >> 5;
    const int n64 = blockIdx.y * 64;
    const bf16* W = Wt + (size_t)4*HDIM*HDIM;        // wo slot
    const int khalf = (lane>>4)<<3, kb16 = (lane>>4)<<4;

    const int trow = blockIdx.x*64 + wid*16 + (lane & 15);   // original token
    const int b = trow >> 11, tl = trow & 2047;
    const int sl = o2s(tl), win = sl >> 6, irow = sl & 63;
    const bf16* crow = CTX + ((size_t)(b*NW + win)*HEADS)*WINSZ*DPH + (size_t)irow*DPH;

    v8f acc[4];
    for (int nt=0;nt<4;++nt) for(int e=0;e<8;++e) acc[nt][e]=0.f;

    for (int kk=0; kk<HDIM; kk+=32) {
        // k = kk + {khalf.., 16+khalf..}: head = kk>>5 fixed, chunks contiguous
        const bf16* hrow = crow + (size_t)(kk>>5)*WINSZ*DPH;
        v16bf a = ld2x8(hrow + ((kk + khalf) & 31), hrow + ((kk + 16 + khalf) & 31));
        #pragma unroll
        for (int nt=0;nt<4;++nt) {
            int nc = n64 + nt*16 + (lane&15);
            v16bf bm = ld2x8(W + (size_t)nc*HDIM + kk + khalf, W + (size_t)nc*HDIM + kk + 16 + khalf);
            acc[nt] = wmma_bf16(a, bm, acc[nt]);
        }
    }
    #pragma unroll
    for (int nt=0;nt<4;++nt) {
        int nc = n64 + nt*16 + (lane&15);
        #pragma unroll
        for (int e=0;e<8;++e) {
            int tr = blockIdx.x*64 + wid*16 + e + ((lane>>4)<<3);
            out[(size_t)tr*HDIM + nc] = acc[nt][e] + bo[nc];
        }
    }
}

// ---------------------------------------------------------------------------
extern "C" void kernel_launch(void* const* d_in, const int* in_sizes, int n_in,
                              void* d_out, int out_size, void* d_ws, size_t ws_size,
                              hipStream_t stream)
{
    const float* x   = (const float*)d_in[0];
    const float* p   = (const float*)d_in[1];
    const float* wq  = (const float*)d_in[2];
    const float* bq  = (const float*)d_in[3];
    const float* wk  = (const float*)d_in[4];
    const float* bk  = (const float*)d_in[5];
    const float* wv  = (const float*)d_in[6];
    const float* bv  = (const float*)d_in[7];
    const float* wo  = (const float*)d_in[8];
    const float* bo  = (const float*)d_in[9];
    const float* p1w = (const float*)d_in[10];
    const float* p1b = (const float*)d_in[11];
    const float* p2w = (const float*)d_in[12];
    const float* p2b = (const float*)d_in[13];
    const float* lng = (const float*)d_in[14];
    const float* lnb = (const float*)d_in[15];
    const float* cv  = (const float*)d_in[16];
    const float* dv  = (const float*)d_in[17];
    float* out = (float*)d_out;

    char* ws = (char*)d_ws;
    const size_t NE = (size_t)NTOK * HDIM;     // 2M elems per buffer
    bf16* Q   = (bf16*)(ws + 0*NE*sizeof(bf16));
    bf16* K   = (bf16*)(ws + 1*NE*sizeof(bf16));
    bf16* V   = (bf16*)(ws + 2*NE*sizeof(bf16));
    bf16* PE  = (bf16*)(ws + 3*NE*sizeof(bf16));
    bf16* Hid = (bf16*)(ws + 4*NE*sizeof(bf16));
    bf16* CTX = (bf16*)(ws + 5*NE*sizeof(bf16));
    bf16* Xs  = (bf16*)(ws + 6*NE*sizeof(bf16));
    bf16* Wt  = (bf16*)(ws + 7*NE*sizeof(bf16));   // 5 x 256x256 bf16 (640KB)

    prep_w_kernel<<<dim3(HDIM, 5), 256, 0, stream>>>(wq, wk, wv, p2w, wo, Wt);
    xshift_kernel<<<NTOK, 256, 0, stream>>>(x, Xs);
    qkv_kernel<<<dim3(NTOK/64, 12), 128, 0, stream>>>(Xs, Wt, bq, bk, bv, Q, K, V);
    posmlp1_kernel<<<NTOK, 256, 0, stream>>>(p, p1w, p1b, lng, lnb, Hid);
    pe_kernel<<<dim3(NTOK/64, 4), 128, 0, stream>>>(Hid, Wt, p2b, PE);
    attn_kernel<<<BATCH*NW*HEADS, 128, 0, stream>>>(Q, K, V, PE, cv, dv, CTX);
    outproj_kernel<<<dim3(NTOK/64, 4), 128, 0, stream>>>(CTX, Wt, bo, out);
    (void)in_sizes; (void)n_in; (void)out_size; (void)ws_size;
}
